// GNN_71811853189872
// MI455X (gfx1250) — matrix operand order repaired and checked by reading, not compiled, not run on hardware
//
#include <hip/hip_runtime.h>
#include <hip/hip_bf16.h>
#include <math.h>

typedef _Float16 v16h __attribute__((ext_vector_type(16)));
typedef _Float16 v8h  __attribute__((ext_vector_type(8)));
typedef float    v8f  __attribute__((ext_vector_type(8)));

#define N_NODES 100000
#define N_EDGES 1600000
#define D_IN    128
#define HIDDEN  128
#define D_OUT   64
#define KDIM    256   // concat [mean | root]

// ---------------- utility kernels ----------------

__global__ void zero_f32(float* __restrict__ p, int n) {
    int tid = blockIdx.x * blockDim.x + threadIdx.x;
    if (tid < n) p[tid] = 0.0f;
}

// one thread per edge: degree count
__global__ void count_edges(const long long* __restrict__ dst, float* __restrict__ cnt, int nE) {
    int e = blockIdx.x * blockDim.x + threadIdx.x;
    if (e >= nE) return;
    unsafeAtomicAdd(cnt + (int)dst[e], 1.0f);
}

// one thread per (edge, 4-feature chunk): agg[dst] += feat[src]
__global__ void scatter_add(const float* __restrict__ feat,
                            const long long* __restrict__ src,
                            const long long* __restrict__ dst,
                            float* __restrict__ agg, int nE) {
    unsigned tid = blockIdx.x * blockDim.x + threadIdx.x;
    unsigned total = (unsigned)nE * 32u;
    if (tid >= total) return;
    int e = tid >> 5;
    int c = (tid & 31) * 4;
    int s = (int)src[e];
    int d = (int)dst[e];
    const float4 v = *(const float4*)(feat + (size_t)s * 128 + c);
    float* base = agg + (size_t)d * 128 + c;
    unsafeAtomicAdd(base + 0, v.x);
    unsafeAtomicAdd(base + 1, v.y);
    unsafeAtomicAdd(base + 2, v.z);
    unsafeAtomicAdd(base + 3, v.w);
}

// Pack A = [mean | root] as f16, row-major [node][256]
__global__ void build_a(const float* __restrict__ agg, const float* __restrict__ cnt,
                        const float* __restrict__ root, _Float16* __restrict__ A, int n) {
    int tid = blockIdx.x * blockDim.x + threadIdx.x;
    if (tid >= n) return;
    int k = tid & 255;
    int node = tid >> 8;
    float v;
    if (k < 128) {
        float c = cnt[node];
        v = agg[(size_t)node * 128 + k] / fmaxf(c, 1.0f);
    } else {
        v = root[(size_t)node * 128 + (k - 128)];
    }
    A[tid] = (_Float16)v;
}

// Pack W^T = [Wl; Wr]^T as f16: Wt[n][k], k = 0..255
__global__ void cvt_w(const float* __restrict__ Wl, const float* __restrict__ Wr,
                      _Float16* __restrict__ Wt, int ncols) {
    int tid = blockIdx.x * blockDim.x + threadIdx.x;
    int total = ncols * 256;
    if (tid >= total) return;
    int k = tid & 255;
    int j = tid >> 8;
    float v = (k < 128) ? Wl[(size_t)k * ncols + j] : Wr[(size_t)(k - 128) * ncols + j];
    Wt[(size_t)j * 256 + k] = (_Float16)v;
}

// ---------------- WMMA GEMM ----------------
// One wave computes a 16x16 output tile: C[mbase:+16, nbase:+16] =
//   act( A[mbase:+16, 0:256] @ Wt[nbase:+16, 0:256]^T + bias[nbase:+16] )
// A layout per ISA (16-bit A 16x32): lane<16 -> K {k..k+7, k+16..k+23},
// lane>=16 -> K {k+8..k+15, k+24..k+31}, row = lane&15.
// B layout (16-bit B 32x16): lane<16 -> K {k..k+7, k+16..k+23}, lane>=16 -> K {k+8..k+15, k+24..k+31}, col = lane&15.
// ACT: 0 = relu, 1 = sigmoid
template <int ACT>
__global__ void gemm_wmma(const _Float16* __restrict__ A,
                          const _Float16* __restrict__ Wt,
                          const float* __restrict__ bias,
                          float* __restrict__ out, int ncols) {
    const int lane = threadIdx.x & 31;
    const int wave = threadIdx.x >> 5;
    const int mbase = blockIdx.x * 16;
    const int nbase = wave * 16;
    const int hs = lane >> 4;        // half-select: 0 or 1
    const int lrow = lane & 15;

    const _Float16* Arow = A  + (size_t)(mbase + lrow) * 256 + hs * 8;
    const _Float16* Brow = Wt + (size_t)(nbase + lrow) * 256 + hs * 8;

    v8f acc = {};
#pragma unroll
    for (int k = 0; k < KDIM; k += 32) {
        v8h alo = *(const v8h*)(Arow + k);
        v8h ahi = *(const v8h*)(Arow + k + 16);
        v16h a = __builtin_shufflevector(alo, ahi, 0,1,2,3,4,5,6,7,8,9,10,11,12,13,14,15);
        v8h blo = *(const v8h*)(Brow + k);
        v8h bhi = *(const v8h*)(Brow + k + 16);
        v16h b = __builtin_shufflevector(blo, bhi, 0,1,2,3,4,5,6,7,8,9,10,11,12,13,14,15);
        acc = __builtin_amdgcn_wmma_f32_16x16x32_f16(
            /*neg_a=*/false, a, /*neg_b=*/false, b,
            /*c_mod=*/(short)0, acc, /*reuse_a=*/false, /*reuse_b=*/false);
    }

    const float bv = bias[nbase + lrow];
#pragma unroll
    for (int r = 0; r < 8; ++r) {
        int row = r + 8 * hs;              // C/D layout: VGPR r -> M=r (lanes 0-15), M=r+8 (lanes 16-31)
        float v = acc[r] + bv;
        if (ACT == 0) {
            v = fmaxf(v, 0.0f);
        } else {
            v = 1.0f / (1.0f + __expf(-v));
        }
        out[(size_t)(mbase + row) * ncols + nbase + lrow] = v;
    }
}

// ---------------- launcher ----------------

extern "C" void kernel_launch(void* const* d_in, const int* in_sizes, int n_in,
                              void* d_out, int out_size, void* d_ws, size_t ws_size,
                              hipStream_t stream) {
    const float*      x    = (const float*)d_in[0];
    const long long*  ei   = (const long long*)d_in[1];   // [2, E] int64
    const float*      Wl1  = (const float*)d_in[2];
    const float*      bl1  = (const float*)d_in[3];
    const float*      Wr1  = (const float*)d_in[4];
    const float*      Wl2  = (const float*)d_in[5];
    const float*      bl2  = (const float*)d_in[6];
    const float*      Wr2  = (const float*)d_in[7];
    float*            outp = (float*)d_out;

    const long long* src = ei;
    const long long* dst = ei + N_EDGES;

    // workspace carve-up
    size_t p = 0;
    auto take = [&](size_t bytes) {
        size_t o = p;
        p += (bytes + 255) & ~(size_t)255;
        return o;
    };
    char* ws = (char*)d_ws;
    float*     cnt = (float*)(ws + take((size_t)N_NODES * 4));
    float*     agg = (float*)(ws + take((size_t)N_NODES * 128 * 4));
    float*     h   = (float*)(ws + take((size_t)N_NODES * 128 * 4));
    _Float16*  A1  = (_Float16*)(ws + take((size_t)N_NODES * 256 * 2));
    _Float16*  W1t = (_Float16*)(ws + take((size_t)128 * 256 * 2));
    _Float16*  W2t = (_Float16*)(ws + take((size_t)64 * 256 * 2));
    (void)ws_size; (void)n_in; (void)in_sizes; (void)out_size;

    const int TB = 256;
    const int aggN = N_NODES * 128;           // 12.8M
    const int a1N  = N_NODES * 256;           // 25.6M
    const unsigned scatterT = (unsigned)N_EDGES * 32u;  // 51.2M

    // ---- layer 1 ----
    zero_f32<<<(N_NODES + TB - 1) / TB, TB, 0, stream>>>(cnt, N_NODES);
    zero_f32<<<(aggN + TB - 1) / TB, TB, 0, stream>>>(agg, aggN);
    count_edges<<<(N_EDGES + TB - 1) / TB, TB, 0, stream>>>(dst, cnt, N_EDGES);
    scatter_add<<<(scatterT + TB - 1) / TB, TB, 0, stream>>>(x, src, dst, agg, N_EDGES);
    build_a<<<(a1N + TB - 1) / TB, TB, 0, stream>>>(agg, cnt, x, A1, a1N);
    cvt_w<<<(128 * 256 + TB - 1) / TB, TB, 0, stream>>>(Wl1, Wr1, W1t, 128);
    cvt_w<<<(64 * 256 + TB - 1) / TB, TB, 0, stream>>>(Wl2, Wr2, W2t, 64);
    // 6250 node tiles of 16 rows; 8 waves cover N=128
    gemm_wmma<0><<<N_NODES / 16, 8 * 32, 0, stream>>>(A1, W1t, bl1, h, 128);

    // ---- layer 2 ----
    zero_f32<<<(aggN + TB - 1) / TB, TB, 0, stream>>>(agg, aggN);
    scatter_add<<<(scatterT + TB - 1) / TB, TB, 0, stream>>>(h, src, dst, agg, N_EDGES);
    build_a<<<(a1N + TB - 1) / TB, TB, 0, stream>>>(agg, cnt, h, A1, a1N);
    // 4 waves cover N=64
    gemm_wmma<1><<<N_NODES / 16, 4 * 32, 0, stream>>>(A1, W2t, bl2, outp, 64);
}
